// DecoderRNN_76390288326890
// MI455X (gfx1250) — matrix operand order repaired and checked by reading, not compiled, not run on hardware
//
#include <hip/hip_runtime.h>

typedef __attribute__((ext_vector_type(16))) __bf16 v16bf;
typedef __attribute__((ext_vector_type(8)))  __bf16 v8bf;
typedef __attribute__((ext_vector_type(4)))  __bf16 v4bf;
typedef __attribute__((ext_vector_type(8)))  float  v8f;
typedef __attribute__((ext_vector_type(4)))  float  v4f;
typedef __attribute__((ext_vector_type(4)))  unsigned v4u;
typedef __attribute__((ext_vector_type(8)))  unsigned v8u;

#define EMBD  512
#define TT    20
#define BB    64
#define VOCAB 32000
#define GATES 2048                // 4*EMBD
#define LDS_STRIDE 520            // bf16 elems per LDS row (512 + 8 pad -> bank-quad skew)

#define WMMA_BF16(a, b, c) \
  __builtin_amdgcn_wmma_f32_16x16x32_bf16(false, (a), false, (b), (short)0, (c), false, false)

__device__ __forceinline__ float sigmoidf_(float x) {
  return 1.0f / (1.0f + __expf(-x));
}

__device__ __forceinline__ void wait_asynccnt0() {
#if __has_builtin(__builtin_amdgcn_s_wait_asynccnt)
  __builtin_amdgcn_s_wait_asynccnt(0);
#else
  asm volatile("s_wait_asynccnt 0" ::: "memory");
#endif
}

__device__ __forceinline__ void wait_tensorcnt0() {
#if __has_builtin(__builtin_amdgcn_s_wait_tensorcnt)
  __builtin_amdgcn_s_wait_tensorcnt(0);
#else
  asm volatile("s_wait_tensorcnt 0" ::: "memory");
#endif
}

// 16-byte global -> LDS async copy (ASYNCcnt tracked, no VGPR data round-trip)
__device__ __forceinline__ void async_g2l_b128(unsigned lds_byte_off, const void* gaddr) {
  asm volatile("global_load_async_to_lds_b128 %0, %1, off"
               :: "v"(lds_byte_off), "v"(gaddr)
               : "memory");
}

// Tensor Data Mover: 2D tile global -> LDS, with per-row LDS padding.
// D# group0: count=1 | lds_addr | 57b global addr | type=2 ("image").
// D# group1: data_size=2B, pad_enable, pad_interval=7 (256 DWORDs = one 1024B row),
//            pad_amount=3 (4 DWORDs = 16B) -> LDS row stride 1040B = LDS_STRIDE bf16.
__device__ __forceinline__ void tdm_load_2d_bf16(unsigned lds_byte_off, const void* gaddr,
                                                 unsigned d0, unsigned d1, unsigned stride0) {
  const unsigned long long ga = (unsigned long long)(uintptr_t)gaddr;
  v4u g0;
  g0[0] = 1u;                                    // count = 1 valid descriptor
  g0[1] = lds_byte_off;                          // lds_addr
  g0[2] = (unsigned)ga;                          // global_addr[31:0]
  g0[3] = (unsigned)(ga >> 32) | (2u << 30);     // global_addr[56:32] | type=2
  v8u g1;
  g1[0] = (1u << 16) | (1u << 20) | (7u << 22) | (3u << 25);  // 2B elems, pad 16B/row
  g1[1] = (d0 & 0xFFFFu) << 16;                  // tensor_dim0[15:0]
  g1[2] = (d0 >> 16) | ((d1 & 0xFFFFu) << 16);   // tensor_dim0[31:16] | tensor_dim1[15:0]
  g1[3] = (d1 >> 16) | ((d0 & 0xFFFFu) << 16);   // tensor_dim1[31:16] | tile_dim0 (= d0)
  g1[4] = d1 & 0xFFFFu;                          // tile_dim1 (= d1), tile_dim2 = 0
  g1[5] = stride0;                               // tensor_dim0_stride[31:0]
  g1[6] = 0u;                                    // stride0 hi | dim1_stride lo
  g1[7] = 0u;
  asm volatile("tensor_load_to_lds %0, %1"
               :: "s"(g0), "s"(g1)
               : "memory");
}

// A operand (16x32 bf16, M x K) from row-major bf16. lanes 0-15: K kh..kh+7 & kh+16..kh+23.
__device__ __forceinline__ v16bf load_a_bf(const __bf16* base, int ld, int row_base,
                                           int kb, int lane) {
  const int m  = lane & 15;
  const int kh = (lane >> 4) * 8;
  const __bf16* p = base + (size_t)(row_base + m) * ld + kb + kh;
  v16bf a;
  *(v8bf*)&a       = *(const v8bf*)p;
  *((v8bf*)&a + 1) = *(const v8bf*)(p + 16);
  return a;
}

// B operand (32x16 bf16, K x N) from row-major W[N][K]: lane = n; lanes 0-15: K kh..kh+7 & kh+16..kh+23.
__device__ __forceinline__ v16bf load_b_bf(const __bf16* W, int ld, int col_base,
                                           int kb, int lane) {
  const int n  = lane & 15;
  const int kh = (lane >> 4) * 8;
  const __bf16* p = W + (size_t)(col_base + n) * ld + kb + kh;
  v16bf b;
  *(v8bf*)&b       = *(const v8bf*)p;
  *((v8bf*)&b + 1) = *(const v8bf*)(p + 16);
  return b;
}

// B operand from f32 row-major W, converting to bf16 in registers.
__device__ __forceinline__ v16bf load_b_f32(const float* rowp, int kb) {
  const float* p = rowp + kb;
  v4f f0 = *(const v4f*)p;
  v4f f1 = *(const v4f*)(p + 4);
  v4f f2 = *(const v4f*)(p + 8);
  v4f f3 = *(const v4f*)(p + 12);
  v16bf b;
#pragma unroll
  for (int i = 0; i < 4; ++i) {
    b[i]      = (__bf16)f0[i];
    b[4 + i]  = (__bf16)f1[i];
    b[8 + i]  = (__bf16)f2[i];
    b[12 + i] = (__bf16)f3[i];
  }
  return b;
}

// ---------------- prep kernels ----------------

__global__ void k_cvt_bf16(const float* __restrict__ src, __bf16* __restrict__ dst, int n4) {
  int i = blockIdx.x * blockDim.x + threadIdx.x;
  if (i < n4) {
    v4f v = ((const v4f*)src)[i];
    v4bf o;
    o[0] = (__bf16)v[0]; o[1] = (__bf16)v[1]; o[2] = (__bf16)v[2]; o[3] = (__bf16)v[3];
    ((v4bf*)dst)[i] = o;
  }
}

__global__ void k_embed(const int* __restrict__ captions, const float* __restrict__ W_emb,
                        __bf16* __restrict__ emb_bf, __bf16* __restrict__ seq_bf) {
  const int b = blockIdx.x / (TT - 1);
  const int t = blockIdx.x % (TT - 1);
  const int idx = captions[b * TT + t];
  const int j = threadIdx.x;  // 128 threads * 4 floats = 512
  v4f v = ((const v4f*)(W_emb + (size_t)idx * EMBD))[j];
  v4bf o;
  o[0] = (__bf16)v[0]; o[1] = (__bf16)v[1]; o[2] = (__bf16)v[2]; o[3] = (__bf16)v[3];
  ((v4bf*)(emb_bf + ((size_t)b * (TT - 1) + t) * EMBD))[j] = o;
  if (t == 0)  // seq[:,0] = emb[:,0]
    ((v4bf*)(seq_bf + (size_t)b * TT * EMBD))[j] = o;
}

__global__ void k_init(const float* __restrict__ features, float* __restrict__ c,
                       __bf16* __restrict__ h0) {
  int i = blockIdx.x * blockDim.x + threadIdx.x;
  v4f v = ((const v4f*)features)[i];
  ((v4f*)c)[i] = v;
  v4bf z;
  z[0] = z[1] = z[2] = z[3] = (__bf16)0.0f;
  ((v4bf*)h0)[i] = z;
}

// ---------------- LSTM step: one wave = 16(batch) x 16(hidden) tile, 4 gate accumulators ----

__global__ void __launch_bounds__(32) k_lstm_step(
    const __bf16* __restrict__ x_base,  // emb_bf + (t-1)*EMBD, row stride (T-1)*EMBD
    const __bf16* __restrict__ h_in,    // [BB][EMBD]
    __bf16* __restrict__ h_out,         // [BB][EMBD]
    float* __restrict__ c,              // [BB][EMBD], in-place
    __bf16* __restrict__ seq_t,         // seq_bf + t*EMBD, row stride T*EMBD
    const __bf16* __restrict__ Wih,     // [GATES][EMBD] bf16
    const __bf16* __restrict__ Whh,     // [GATES][EMBD] bf16
    const float* __restrict__ b_ih, const float* __restrict__ b_hh) {
  const int lane = threadIdx.x;
  const int b0 = (blockIdx.x >> 5) * 16;
  const int e0 = (blockIdx.x & 31) * 16;

  v8f acc[4];
#pragma unroll
  for (int q = 0; q < 4; ++q)
#pragma unroll
    for (int r = 0; r < 8; ++r) acc[q][r] = 0.0f;

  for (int kb = 0; kb < EMBD; kb += 32) {
    v16bf ax = load_a_bf(x_base, (TT - 1) * EMBD, b0, kb, lane);
    v16bf ah = load_a_bf(h_in, EMBD, b0, kb, lane);
#pragma unroll
    for (int q = 0; q < 4; ++q) {
      v16bf bi = load_b_bf(Wih, EMBD, q * EMBD + e0, kb, lane);
      acc[q] = WMMA_BF16(ax, bi, acc[q]);
      v16bf bh = load_b_bf(Whh, EMBD, q * EMBD + e0, kb, lane);
      acc[q] = WMMA_BF16(ah, bh, acc[q]);
    }
  }

  const int n  = lane & 15;
  const int mh = (lane >> 4) * 8;
  const float bsi = b_ih[0 * EMBD + e0 + n] + b_hh[0 * EMBD + e0 + n];
  const float bsf = b_ih[1 * EMBD + e0 + n] + b_hh[1 * EMBD + e0 + n];
  const float bsg = b_ih[2 * EMBD + e0 + n] + b_hh[2 * EMBD + e0 + n];
  const float bso = b_ih[3 * EMBD + e0 + n] + b_hh[3 * EMBD + e0 + n];

#pragma unroll
  for (int r = 0; r < 8; ++r) {
    const int bb = b0 + mh + r;
    const size_t off = (size_t)bb * EMBD + e0 + n;
    float iv = sigmoidf_(acc[0][r] + bsi);
    float fv = sigmoidf_(acc[1][r] + bsf);
    float gv = tanhf(acc[2][r] + bsg);
    float ov = sigmoidf_(acc[3][r] + bso);
    float cn = fv * c[off] + iv * gv;
    c[off] = cn;
    __bf16 hb = (__bf16)(ov * tanhf(cn));
    h_out[off] = hb;
    seq_t[(size_t)bb * TT * EMBD + e0 + n] = hb;
  }
}

// ---------------- output projection ----------------
// Block = 8 waves sharing one 64-row A panel, staged by ONE tensor_load_to_lds (TDM DMA,
// hardware row padding for LDS bank skew). Wave 0 also async-stages the block's b_out
// slice into LDS. K-loop: A from LDS (ds pipe), W_out streamed f32->bf16 (vmem pipe),
// WMMA on the XDL pipe.

__global__ void __launch_bounds__(256) k_logits(const __bf16* __restrict__ seq_bf,  // [1280][512]
                                                const float* __restrict__ W_out,    // [32000][512]
                                                const float* __restrict__ b_out,
                                                float* __restrict__ out) {          // [1280][32000]
  __shared__ __bf16 apanel[64 * LDS_STRIDE];  // 66,560 B (padded rows, conflict-free b128)
  __shared__ float  bias_sm[128];

  const int tid  = threadIdx.x;
  const int lane = tid & 31;
  const int wv   = tid >> 5;                 // 0..7
  const int mg   = blockIdx.x / 250;         // 20 m-groups of 64 rows
  const int ng   = blockIdx.x % 250;         // 250 n-groups of 128 cols
  const int m0   = mg * 64;
  const int n0   = ng * 128 + wv * 16;

  // ---- wave 0: TDM-stage A panel (64 x 512 bf16) + async-stage 128-float bias slice ----
  if (wv == 0) {
    tdm_load_2d_bf16((unsigned)(uintptr_t)apanel,
                     seq_bf + (size_t)m0 * EMBD,
                     /*d0=*/EMBD, /*d1=*/64, /*stride0=*/EMBD);
    async_g2l_b128((unsigned)(uintptr_t)bias_sm + (unsigned)lane * 16,
                   (const char*)(b_out + ng * 128) + lane * 16);
  }
  wait_tensorcnt0();
  wait_asynccnt0();
  __syncthreads();

  // this lane's streamed W_out row chunk (f32, converted on the fly)
  const float* wrow = W_out + (size_t)(n0 + (lane & 15)) * EMBD + (lane >> 4) * 16;

  v8f acc[4];
#pragma unroll
  for (int s = 0; s < 4; ++s)
#pragma unroll
    for (int r = 0; r < 8; ++r) acc[s][r] = 0.0f;

  const int am  = lane & 15;
  const int akh = (lane >> 4) * 16;

#pragma unroll 2
  for (int kb = 0; kb < EMBD; kb += 32) {
    if (kb + 32 < EMBD) __builtin_prefetch(wrow + kb + 32, 0, 1);
    v16bf bt = load_b_f32(wrow, kb);
#pragma unroll
    for (int s = 0; s < 4; ++s) {
      const __bf16* p = apanel + (size_t)(s * 16 + am) * LDS_STRIDE + kb + akh;
      v16bf a;
      *(v8bf*)&a       = *(const v8bf*)p;         // ds_load_b128
      *((v8bf*)&a + 1) = *(const v8bf*)(p + 8);   // ds_load_b128
      acc[s] = WMMA_BF16(a, bt, acc[s]);
    }
  }

  const int n  = n0 + (lane & 15);
  const int mh = (lane >> 4) * 8;
  const float bo = bias_sm[wv * 16 + (lane & 15)];
#pragma unroll
  for (int s = 0; s < 4; ++s)
#pragma unroll
    for (int r = 0; r < 8; ++r) {
      const int m = m0 + s * 16 + mh + r;
      out[(size_t)m * VOCAB + n] = acc[s][r] + bo;
    }
}

// ---------------- host launcher ----------------

extern "C" void kernel_launch(void* const* d_in, const int* in_sizes, int n_in,
                              void* d_out, int out_size, void* d_ws, size_t ws_size,
                              hipStream_t stream) {
  const float* features = (const float*)d_in[0];   // [64,512]
  const int*   captions = (const int*)d_in[1];     // [64,20]
  const float* W_emb    = (const float*)d_in[2];   // [32000,512]
  const float* W_out    = (const float*)d_in[3];   // [32000,512]
  const float* b_out    = (const float*)d_in[4];   // [32000]
  const float* W_ih     = (const float*)d_in[5];   // [2048,512]
  const float* W_hh     = (const float*)d_in[6];   // [2048,512]
  const float* b_ih     = (const float*)d_in[7];   // [2048]
  const float* b_hh     = (const float*)d_in[8];   // [2048]
  float* out = (float*)d_out;

  char* w = (char*)d_ws;
  __bf16* Wih_bf = (__bf16*)w; w += (size_t)GATES * EMBD * 2;
  __bf16* Whh_bf = (__bf16*)w; w += (size_t)GATES * EMBD * 2;
  __bf16* emb_bf = (__bf16*)w; w += (size_t)BB * (TT - 1) * EMBD * 2;
  __bf16* seq_bf = (__bf16*)w; w += (size_t)BB * TT * EMBD * 2;
  __bf16* h_bf   = (__bf16*)w; w += (size_t)2 * BB * EMBD * 2;
  float*  c_buf  = (float*)w;  w += (size_t)BB * EMBD * 4;

  k_cvt_bf16<<<1024, 256, 0, stream>>>(W_ih, Wih_bf, GATES * EMBD / 4);
  k_cvt_bf16<<<1024, 256, 0, stream>>>(W_hh, Whh_bf, GATES * EMBD / 4);
  k_embed<<<BB * (TT - 1), 128, 0, stream>>>(captions, W_emb, emb_bf, seq_bf);
  k_init<<<(BB * EMBD / 4) / 256, 256, 0, stream>>>(features, c_buf, h_bf);

  for (int t = 1; t < TT; ++t) {
    const __bf16* h_in = h_bf + (size_t)((t - 1) & 1) * BB * EMBD;
    __bf16* h_out      = h_bf + (size_t)(t & 1) * BB * EMBD;
    k_lstm_step<<<128, 32, 0, stream>>>(emb_bf + (size_t)(t - 1) * EMBD, h_in, h_out,
                                        c_buf, seq_bf + (size_t)t * EMBD,
                                        Wih_bf, Whh_bf, b_ih, b_hh);
  }

  // 1280 x 32000: 20 m-groups x 250 n-groups = 5000 blocks x 8 waves
  k_logits<<<5000, 256, 0, stream>>>(seq_bf, W_out, b_out, out);
}
